// StockTransformerModel_49890340111096
// MI455X (gfx1250) — matrix operand order repaired and checked by reading, not compiled, not run on hardware
//
#include <hip/hip_runtime.h>
#include <math.h>
#include <stdint.h>

// ---------------------------------------------------------------------------
// Model constants (from reference)
// ---------------------------------------------------------------------------
namespace {
constexpr int Lc = 4, Bc = 4, Sc = 256, Dc = 1024, Hc = 16;
constexpr int DNc = 64, DRc = 32, DVc = 64;
constexpr int RQc = 384, RKVc = 256;
constexpr int Ec = 8, Mc = 512, MSc = 2048;
constexpr int FINc = 64;
constexpr int Tc = Bc * Sc;          // 1024 tokens
constexpr int QHc = DNc + DRc;       // 96  (per-head q/k dim)
constexpr int KVHc = DNc + DVc;      // 128 (per-head kv dim)
}

typedef __attribute__((ext_vector_type(16))) _Float16 v16h;
typedef __attribute__((ext_vector_type(8)))  _Float16 v8h;
typedef __attribute__((ext_vector_type(4)))  _Float16 v4h;
typedef __attribute__((ext_vector_type(8)))  float    v8f;
typedef __attribute__((ext_vector_type(4)))  float    v4f;
typedef int v2i __attribute__((vector_size(8)));   // matches builtin proto

// Async global->LDS path (gfx1250).  Guarded so an absent builtin falls back
// to a synchronous vector copy instead of breaking the build.
#if defined(__gfx1250__) && __has_builtin(__builtin_amdgcn_global_load_async_to_lds_b64)
#define ASYNC_LDS 1
#else
#define ASYNC_LDS 0
#endif

// ---------------------------------------------------------------------------
// Shared fragment helpers (ISA 7.12.2 wave32 layouts)
// ---------------------------------------------------------------------------
__device__ __forceinline__ v16h frag_cat(v8h lo, v8h hi) {
  return __builtin_shufflevector(lo, hi, 0,1,2,3,4,5,6,7,8,9,10,11,12,13,14,15);
}

// ---------------------------------------------------------------------------
// WMMA GEMM, f32 B operand (activation x activation):
//   C[M,N] = A[M,K] @ B          TRANSB: B is [N,K] row-major
// M % 128 == 0, N % 32 == 0, K % 32 == 0, strides multiple of 4 floats.
// Block = 256 thr (8 waves), tile 128x32, K-step 32, 2 WMMA / iter / wave.
// Batched via grid.z with split (outer,inner) strides.
// ---------------------------------------------------------------------------
template <int TRANSB>
__global__ __launch_bounds__(256)
void gemm_wmma(const float* __restrict__ A, const float* __restrict__ B,
               float* __restrict__ C,
               int Kd, int lda, int ldb, int ldc,
               int batchH,
               long sAb, long sAh, long sBb, long sBh, long sCb, long sCh)
{
  __shared__ __align__(16) _Float16 Al[128 * 32];  // [row][k]
  __shared__ __align__(16) _Float16 Bl[32 * 32];   // [n][k]

  const int z  = blockIdx.z;
  const int zb = z / batchH;
  const int zh = z - zb * batchH;
  A += zb * sAb + zh * sAh;
  B += zb * sBb + zh * sBh;
  C += zb * sCb + zh * sCh;

  const int m0 = blockIdx.y * 128;
  const int n0 = blockIdx.x * 32;

  const int tid  = threadIdx.x;
  const int wave = tid >> 5;
  const int lane = tid & 31;
  const int lrow = lane & 15;
  const int half = lane >> 4;

  const int ar = tid >> 1;              // A staging: 2 thr/row, 16 elem each
  const int ak = (tid & 1) * 16;
  const float* Ag = A + (long)(m0 + ar) * lda + ak;

  const float* Bg;
  int bn, bk;
  if (TRANSB) { bn = tid >> 3; bk = (tid & 7) * 4; Bg = B + (long)(n0 + bn) * ldb + bk; }
  else        { bk = tid >> 3; bn = (tid & 7) * 4; Bg = B + (long)bk * ldb + n0 + bn; }
  const long bAdv = TRANSB ? 32 : (long)32 * ldb;

  v8f acc0 = {}, acc1 = {};

  for (int k0 = 0; k0 < Kd; k0 += 32) {
    if (k0 + 32 < Kd) {
      __builtin_prefetch(Ag + 32, 0, 1);
      __builtin_prefetch(Bg + bAdv, 0, 1);
    }
    {   // stage A: 4 x b128 -> cvt -> 2 x ds b128
      const v4f* p = (const v4f*)Ag;
      v4f f0 = p[0], f1 = p[1], f2 = p[2], f3 = p[3];
      v8h h0, h1;
#pragma unroll
      for (int j = 0; j < 4; ++j) {
        h0[j] = (_Float16)f0[j]; h0[j + 4] = (_Float16)f1[j];
        h1[j] = (_Float16)f2[j]; h1[j + 4] = (_Float16)f3[j];
      }
      *(v8h*)&Al[ar * 32 + ak]     = h0;
      *(v8h*)&Al[ar * 32 + ak + 8] = h1;
    }
    {   // stage B
      v4f f = *(const v4f*)Bg;
      if (TRANSB) {
        v4h hb;
#pragma unroll
        for (int j = 0; j < 4; ++j) hb[j] = (_Float16)f[j];
        *(v4h*)&Bl[bn * 32 + bk] = hb;
      } else {
#pragma unroll
        for (int j = 0; j < 4; ++j) Bl[(bn + j) * 32 + bk] = (_Float16)f[j];
      }
    }
    __syncthreads();

    const int arow = wave * 16 + lrow;
    v16h a  = frag_cat(*(const v8h*)&Al[arow * 32 + half * 8],
                       *(const v8h*)&Al[arow * 32 + 16 + half * 8]);
    v16h b0 = frag_cat(*(const v8h*)&Bl[lrow * 32 + half * 8],
                       *(const v8h*)&Bl[lrow * 32 + 16 + half * 8]);
    v16h b1 = frag_cat(*(const v8h*)&Bl[(16 + lrow) * 32 + half * 8],
                       *(const v8h*)&Bl[(16 + lrow) * 32 + 16 + half * 8]);

    acc0 = __builtin_amdgcn_wmma_f32_16x16x32_f16(false, a, false, b0, (short)0,
                                                  acc0, false, false);
    acc1 = __builtin_amdgcn_wmma_f32_16x16x32_f16(false, a, false, b1, (short)0,
                                                  acc1, false, false);
    __syncthreads();

    Ag += 32;
    Bg += bAdv;
  }

  const int rbase = m0 + wave * 16 + half * 8;
  const int col   = n0 + lrow;
#pragma unroll
  for (int r = 0; r < 8; ++r) {
    long i0 = (long)(rbase + r) * ldc + col;
    C[i0]      = acc0[r];
    C[i0 + 16] = acc1[r];
  }
}

// ---------------------------------------------------------------------------
// WMMA GEMM, pre-transposed f16 weights:  C[M,N] (+)= A[M,K] @ Bt^T
// Bt is f16 [N,K] row-major (ldb = K), so B tiles are contiguous along K and
// are copied global->LDS with GLOBAL_LOAD_ASYNC_TO_LDS_B64 (ASYNCcnt path,
// no VGPR round-trip, half the bytes of the f32 path).
// ---------------------------------------------------------------------------
template <int ACC>
__global__ __launch_bounds__(256)
void gemm_wmma_wt(const float* __restrict__ A, const _Float16* __restrict__ Bt,
                  float* __restrict__ C, int Kd, int lda, int ldb, int ldc)
{
  __shared__ __align__(16) _Float16 Al[128 * 32];  // [row][k]
  __shared__ __align__(16) _Float16 Bl[32 * 32];   // [n][k]

  const int m0 = blockIdx.y * 128;
  const int n0 = blockIdx.x * 32;

  const int tid  = threadIdx.x;
  const int wave = tid >> 5;
  const int lane = tid & 31;
  const int lrow = lane & 15;
  const int half = lane >> 4;

  const int ar = tid >> 1;
  const int ak = (tid & 1) * 16;
  const float* Ag = A + (long)(m0 + ar) * lda + ak;

  const int bn = tid >> 3;          // 0..31
  const int bk = (tid & 7) * 4;     // 0..28 (4 halves = 8B)
  const _Float16* Bg = Bt + (long)(n0 + bn) * ldb + bk;

  v8f acc0 = {}, acc1 = {};

  for (int k0 = 0; k0 < Kd; k0 += 32) {
    if (k0 + 32 < Kd) {
      __builtin_prefetch(Ag + 32, 0, 1);
      __builtin_prefetch(Bg + 32, 0, 1);
    }
    {   // stage A: 4 x b128 -> cvt -> 2 x ds b128
      const v4f* p = (const v4f*)Ag;
      v4f f0 = p[0], f1 = p[1], f2 = p[2], f3 = p[3];
      v8h h0, h1;
#pragma unroll
      for (int j = 0; j < 4; ++j) {
        h0[j] = (_Float16)f0[j]; h0[j + 4] = (_Float16)f1[j];
        h1[j] = (_Float16)f2[j]; h1[j + 4] = (_Float16)f3[j];
      }
      *(v8h*)&Al[ar * 32 + ak]     = h0;
      *(v8h*)&Al[ar * 32 + ak + 8] = h1;
    }
    // stage B: direct async global->LDS copy (8B per lane)
#if ASYNC_LDS
    __builtin_amdgcn_global_load_async_to_lds_b64(
        (__attribute__((address_space(1))) v2i*)Bg,
        (__attribute__((address_space(3))) v2i*)&Bl[bn * 32 + bk], 0, 0);
    asm volatile("s_wait_asynccnt 0" ::: "memory");
#else
    *(v4h*)&Bl[bn * 32 + bk] = *(const v4h*)Bg;
#endif
    __syncthreads();

    const int arow = wave * 16 + lrow;
    v16h a  = frag_cat(*(const v8h*)&Al[arow * 32 + half * 8],
                       *(const v8h*)&Al[arow * 32 + 16 + half * 8]);
    v16h b0 = frag_cat(*(const v8h*)&Bl[lrow * 32 + half * 8],
                       *(const v8h*)&Bl[lrow * 32 + 16 + half * 8]);
    v16h b1 = frag_cat(*(const v8h*)&Bl[(16 + lrow) * 32 + half * 8],
                       *(const v8h*)&Bl[(16 + lrow) * 32 + 16 + half * 8]);

    acc0 = __builtin_amdgcn_wmma_f32_16x16x32_f16(false, a, false, b0, (short)0,
                                                  acc0, false, false);
    acc1 = __builtin_amdgcn_wmma_f32_16x16x32_f16(false, a, false, b1, (short)0,
                                                  acc1, false, false);
    __syncthreads();

    Ag += 32;
    Bg += 32;
  }

  const int rbase = m0 + wave * 16 + half * 8;
  const int col   = n0 + lrow;
#pragma unroll
  for (int r = 0; r < 8; ++r) {
    long i0 = (long)(rbase + r) * ldc + col;
    long i1 = i0 + 16;
    if (ACC) { C[i0] += acc0[r]; C[i1] += acc1[r]; }
    else     { C[i0]  = acc0[r]; C[i1]  = acc1[r]; }
  }
}

// ---------------------------------------------------------------------------
// One-time weight convert+transpose: Wt[mat][n][k] = (f16) W[mat][k][n]
// ---------------------------------------------------------------------------
__global__ __launch_bounds__(256)
void transpose_cvt_kernel(const float* __restrict__ W, _Float16* __restrict__ Wt,
                          int K, int N, long total)
{
  long idx = (long)blockIdx.x * 256 + threadIdx.x;
  if (idx >= total) return;
  long mk  = (long)K * N;
  long mat = idx / mk;
  long rem = idx - mat * mk;
  int  n   = (int)(rem / K);
  int  k   = (int)(rem - (long)n * K);
  Wt[idx] = (_Float16)W[mat * mk + (long)k * N + n];
}

// ---------------------------------------------------------------------------
// Skinny GEMM for N <= 8 (router logits, final head).
// ---------------------------------------------------------------------------
__global__ __launch_bounds__(256)
void matvec_small(const float* __restrict__ A, const float* __restrict__ B,
                  float* __restrict__ C, int N, int K, int lda, int ldb, int ldc)
{
  __shared__ float red[256];
  const int m = blockIdx.x;
  const int tid = threadIdx.x;
  for (int n = 0; n < N; ++n) {
    float s = 0.0f;
    for (int k = tid; k < K; k += 256)
      s += A[(long)m * lda + k] * B[(long)k * ldb + n];
    red[tid] = s;
    __syncthreads();
    for (int off = 128; off > 0; off >>= 1) {
      if (tid < off) red[tid] += red[tid + off];
      __syncthreads();
    }
    if (tid == 0) C[(long)m * ldc + n] = red[0];
    __syncthreads();
  }
}

// ---------------------------------------------------------------------------
// RMSNorm: one block (256 thr) per row.
// ---------------------------------------------------------------------------
__global__ __launch_bounds__(256)
void rms_kernel(const float* __restrict__ x, const float* __restrict__ w,
                float* __restrict__ y, int n, int ldx, int ldy)
{
  __shared__ float red[256];
  const int row = blockIdx.x;
  const int tid = threadIdx.x;
  const float* xr = x + (long)row * ldx;
  float s = 0.0f;
  for (int i = tid; i < n; i += 256) { float v = xr[i]; s += v * v; }
  red[tid] = s;
  __syncthreads();
  for (int off = 128; off > 0; off >>= 1) {
    if (tid < off) red[tid] += red[tid + off];
    __syncthreads();
  }
  const float inv = rsqrtf(red[0] / (float)n + 1e-6f);
  float* yr = y + (long)row * ldy;
  for (int i = tid; i < n; i += 256) yr[i] = xr[i] * inv * w[i];
}

// ---------------------------------------------------------------------------
// RoPE on q[..., DN:DN+DR] in-place.  q layout (B,S,H,QH).
// ---------------------------------------------------------------------------
__global__ __launch_bounds__(256)
void rope_q_kernel(float* __restrict__ q, const float* __restrict__ cosb,
                   const float* __restrict__ sinb, int total)
{
  int idx = blockIdx.x * 256 + threadIdx.x;
  if (idx >= total) return;
  const int P = DRc / 2;
  int i = idx % P;
  int h = (idx / P) % Hc;
  int s = (idx / (P * Hc)) % Sc;
  int b = idx / (P * Hc * Sc);
  float c  = cosb[s * P + i];
  float sn = sinb[s * P + i];
  float* p = q + ((long)((b * Sc + s) * Hc + h) * QHc + DNc + 2 * i);
  float a = p[0], bb = p[1];
  p[0] = a * c - bb * sn;
  p[1] = a * sn + bb * c;
}

// ---------------------------------------------------------------------------
// Build Keff (B,S,H,QH): [kn | roped k_pe broadcast over heads]
// ---------------------------------------------------------------------------
__global__ __launch_bounds__(256)
void build_keff_kernel(const float* __restrict__ kv, const float* __restrict__ kva,
                       const float* __restrict__ cosb, const float* __restrict__ sinb,
                       float* __restrict__ keff, int total)
{
  int idx = blockIdx.x * 256 + threadIdx.x;
  if (idx >= total) return;
  int d = idx % QHc;
  int h = (idx / QHc) % Hc;
  int s = (idx / (QHc * Hc)) % Sc;
  int b = idx / (QHc * Hc * Sc);
  float out;
  if (d < DNc) {
    out = kv[(long)((b * Sc + s) * Hc + h) * KVHc + d];
  } else {
    int i = (d - DNc) >> 1;
    const float* kp = kva + (long)(b * Sc + s) * (RKVc + DRc) + RKVc;
    float a = kp[2 * i], bb = kp[2 * i + 1];
    float c = cosb[s * (DRc / 2) + i], sn = sinb[s * (DRc / 2) + i];
    out = ((d & 1) == 0) ? (a * c - bb * sn) : (a * sn + bb * c);
  }
  keff[idx] = out;
}

// ---------------------------------------------------------------------------
// Causal scaled softmax over scores (B*H, S, S), in place.
// ---------------------------------------------------------------------------
__global__ __launch_bounds__(256)
void softmax_kernel(float* __restrict__ scores)
{
  __shared__ float red[256];
  const int z  = blockIdx.x;
  const int q  = z % Sc;
  const int bh = z / Sc;
  float* row = scores + ((long)bh * Sc + q) * Sc;
  const int k = threadIdx.x;
  const float SCALE = rsqrtf((float)(DNc + DRc));
  float v = (k <= q) ? row[k] * SCALE : -1.0e9f;
  red[k] = v;
  __syncthreads();
  for (int off = 128; off > 0; off >>= 1) {
    if (k < off) red[k] = fmaxf(red[k], red[k + off]);
    __syncthreads();
  }
  const float m = red[0];
  __syncthreads();
  float e = expf(v - m);
  red[k] = e;
  __syncthreads();
  for (int off = 128; off > 0; off >>= 1) {
    if (k < off) red[k] += red[k + off];
    __syncthreads();
  }
  row[k] = e / red[0];
}

// ---------------------------------------------------------------------------
// Router: softmax over E=8, top-2, renormalized dense gate.
// ---------------------------------------------------------------------------
__global__ __launch_bounds__(256)
void router_kernel(const float* __restrict__ rlog, float* __restrict__ gate, int T)
{
  int t = blockIdx.x * 256 + threadIdx.x;
  if (t >= T) return;
  const float* lg = rlog + (long)t * Ec;
  float m = lg[0];
  for (int e = 1; e < Ec; ++e) m = fmaxf(m, lg[e]);
  float p[Ec]; float sum = 0.0f;
  for (int e = 0; e < Ec; ++e) { p[e] = expf(lg[e] - m); sum += p[e]; }
  for (int e = 0; e < Ec; ++e) p[e] /= sum;
  int i1 = 0; for (int e = 1; e < Ec; ++e) if (p[e] > p[i1]) i1 = e;
  int i2 = (i1 == 0) ? 1 : 0;
  for (int e = 0; e < Ec; ++e) if (e != i1 && p[e] > p[i2]) i2 = e;
  float w = p[i1] + p[i2];
  float* g = gate + (long)t * Ec;
  for (int e = 0; e < Ec; ++e) g[e] = 0.0f;
  g[i1] = p[i1] / w;
  g[i2] = p[i2] / w;
}

// ---------------------------------------------------------------------------
// u = silu(g1) * g3 [* gate[t][e] if gate != nullptr]
// ---------------------------------------------------------------------------
__global__ __launch_bounds__(256)
void silu_mul_kernel(const float* __restrict__ g1, const float* __restrict__ g3,
                     float* __restrict__ u, const float* __restrict__ gate,
                     int e, int n, int total)
{
  int idx = blockIdx.x * 256 + threadIdx.x;
  if (idx >= total) return;
  float x = g1[idx];
  float v = (x / (1.0f + expf(-x))) * g3[idx];
  if (gate) {
    int t = idx / n;
    v *= gate[(long)t * Ec + e];
  }
  u[idx] = v;
}

// ---------------------------------------------------------------------------
// Host-side orchestration
// ---------------------------------------------------------------------------
static inline void launch_gemm_f32B(hipStream_t stream,
                                    const float* A, const float* B, float* C,
                                    int M, int N, int K, int lda, int ldb, int ldc,
                                    int trB, int batch, int batchH,
                                    long sAb, long sAh, long sBb, long sBh,
                                    long sCb, long sCh)
{
  dim3 grid(N / 32, M / 128, batch);
  if (trB) gemm_wmma<1><<<grid, 256, 0, stream>>>(A,B,C,K,lda,ldb,ldc,batchH,sAb,sAh,sBb,sBh,sCb,sCh);
  else     gemm_wmma<0><<<grid, 256, 0, stream>>>(A,B,C,K,lda,ldb,ldc,batchH,sAb,sAh,sBb,sBh,sCb,sCh);
}

static inline void launch_gemm_wt(hipStream_t stream,
                                  const float* A, const _Float16* Bt, float* C,
                                  int M, int N, int K, int lda, int ldc, int acc)
{
  dim3 grid(N / 32, M / 128, 1);
  if (acc) gemm_wmma_wt<1><<<grid, 256, 0, stream>>>(A, Bt, C, K, lda, K, ldc);
  else     gemm_wmma_wt<0><<<grid, 256, 0, stream>>>(A, Bt, C, K, lda, K, ldc);
}

extern "C" void kernel_launch(void* const* d_in, const int* in_sizes, int n_in,
                              void* d_out, int out_size, void* d_ws, size_t ws_size,
                              hipStream_t stream)
{
  (void)in_sizes; (void)n_in; (void)out_size; (void)ws_size;

  const float* x        = (const float*)d_in[0];   // (B,S,FIN)
  const float* rope_cos = (const float*)d_in[1];   // (S,16)
  const float* rope_sin = (const float*)d_in[2];   // (S,16)
  const float* W_in     = (const float*)d_in[3];   // (FIN,D)
  const float* attn_nw  = (const float*)d_in[4];   // (L,D)
  const float* Wqa      = (const float*)d_in[5];   // (L,D,RQ)
  const float* q_nw     = (const float*)d_in[6];   // (L,RQ)
  const float* Wqb      = (const float*)d_in[7];   // (L,RQ,H*QH)
  const float* Wkva     = (const float*)d_in[8];   // (L,D,RKV+DR)
  const float* kv_nw    = (const float*)d_in[9];   // (L,RKV)
  const float* Wkvb     = (const float*)d_in[10];  // (L,RKV,H*KVH)
  const float* Wo       = (const float*)d_in[11];  // (L,H*DV,D)
  const float* moe_nw   = (const float*)d_in[12];  // (L,D)
  const float* Wr       = (const float*)d_in[13];  // (L,D,E)
  const float* We1      = (const float*)d_in[14];  // (L,E,D,M)
  const float* We3      = (const float*)d_in[15];  // (L,E,D,M)
  const float* We2      = (const float*)d_in[16];  // (L,E,M,D)
  const float* Ws1      = (const float*)d_in[17];  // (L,D,MS)
  const float* Ws3      = (const float*)d_in[18];  // (L,D,MS)
  const float* Ws2      = (const float*)d_in[19];  // (L,MS,D)
  const float* final_nw = (const float*)d_in[20];  // (D)
  const float* W_out    = (const float*)d_in[21];  // (D,1)
  float* out = (float*)d_out;                      // (B,S,1) f32

  const int KVA = RKVc + DRc;  // 288
  const int QW  = Hc * QHc;    // 1536
  const int KVW = Hc * KVHc;   // 2048

  // ---- workspace: f32 activations ----
  float* ws   = (float*)d_ws;
  float* hs   = ws;                            // T x D
  float* h    = hs  + (long)Tc * Dc;           // T x D
  float* qa   = h   + (long)Tc * Dc;           // T x RQ
  float* qb   = qa  + (long)Tc * RQc;          // T x QW
  float* kva  = qb  + (long)Tc * QW;           // T x KVA
  float* kv   = kva + (long)Tc * KVA;          // T x KVW
  float* keff = kv  + (long)Tc * KVW;          // T x QW
  float* sc   = keff + (long)Tc * QW;          // B*H x S x S
  float* ao   = sc  + (long)Bc * Hc * Sc * Sc; // T x H*DV
  float* g1   = ao  + (long)Tc * Hc * DVc;     // T x MS
  float* g3   = g1  + (long)Tc * MSc;          // T x MS
  float* u    = g3  + (long)Tc * MSc;          // T x MS
  float* rlog = u   + (long)Tc * MSc;          // T x E
  float* gate = rlog + (long)Tc * Ec;          // T x E

  // ---- workspace: f16 transposed weights ([N][K] per matrix) ----
  _Float16* wf = (_Float16*)((((uintptr_t)(gate + (long)Tc * Ec)) + 255) & ~(uintptr_t)255);
  _Float16* tW_in = wf;                                  // [D][FIN]
  _Float16* tWqa  = tW_in + (long)FINc * Dc;             // L x [RQ][D]
  _Float16* tWqb  = tWqa  + (long)Lc * Dc * RQc;         // L x [QW][RQ]
  _Float16* tWkva = tWqb  + (long)Lc * RQc * QW;         // L x [KVA][D]
  _Float16* tWkvb = tWkva + (long)Lc * Dc * KVA;         // L x [KVW][RKV]
  _Float16* tWo   = tWkvb + (long)Lc * RKVc * KVW;       // L x [D][H*DV]
  _Float16* tWe1  = tWo   + (long)Lc * Dc * Dc;          // L*E x [M][D]
  _Float16* tWe3  = tWe1  + (long)Lc * Ec * Dc * Mc;     // L*E x [M][D]
  _Float16* tWe2  = tWe3  + (long)Lc * Ec * Dc * Mc;     // L*E x [D][M]
  _Float16* tWs1  = tWe2  + (long)Lc * Ec * Mc * Dc;     // L x [MS][D]
  _Float16* tWs3  = tWs1  + (long)Lc * Dc * MSc;         // L x [MS][D]
  _Float16* tWs2  = tWs3  + (long)Lc * Dc * MSc;         // L x [D][MS]

  auto tcvt = [&](const float* W, _Float16* Wt, int K, int N, int nmat) {
    long total = (long)nmat * K * N;
    transpose_cvt_kernel<<<(int)((total + 255) / 256), 256, 0, stream>>>(W, Wt, K, N, total);
  };
  tcvt(W_in, tW_in, FINc, Dc, 1);
  tcvt(Wqa,  tWqa,  Dc,   RQc, Lc);
  tcvt(Wqb,  tWqb,  RQc,  QW,  Lc);
  tcvt(Wkva, tWkva, Dc,   KVA, Lc);
  tcvt(Wkvb, tWkvb, RKVc, KVW, Lc);
  tcvt(Wo,   tWo,   Dc,   Dc,  Lc);
  tcvt(We1,  tWe1,  Dc,   Mc,  Lc * Ec);
  tcvt(We3,  tWe3,  Dc,   Mc,  Lc * Ec);
  tcvt(We2,  tWe2,  Mc,   Dc,  Lc * Ec);
  tcvt(Ws1,  tWs1,  Dc,   MSc, Lc);
  tcvt(Ws3,  tWs3,  Dc,   MSc, Lc);
  tcvt(Ws2,  tWs2,  MSc,  Dc,  Lc);

  // hs = x @ W_in
  launch_gemm_wt(stream, x, tW_in, hs, Tc, Dc, FINc, FINc, Dc, 0);

  for (int l = 0; l < Lc; ++l) {
    // h = rms(hs, attn_norm_w[l])
    rms_kernel<<<Tc, 256, 0, stream>>>(hs, attn_nw + (long)l * Dc, h, Dc, Dc, Dc);

    // q path
    launch_gemm_wt(stream, h, tWqa + (long)l * Dc * RQc, qa, Tc, RQc, Dc, Dc, RQc, 0);
    rms_kernel<<<Tc, 256, 0, stream>>>(qa, q_nw + (long)l * RQc, qa, RQc, RQc, RQc);
    launch_gemm_wt(stream, qa, tWqb + (long)l * RQc * QW, qb, Tc, QW, RQc, RQc, QW, 0);

    // kv path
    launch_gemm_wt(stream, h, tWkva + (long)l * Dc * KVA, kva, Tc, KVA, Dc, Dc, KVA, 0);
    rms_kernel<<<Tc, 256, 0, stream>>>(kva, kv_nw + (long)l * RKVc, kva, RKVc, KVA, KVA);
    launch_gemm_wt(stream, kva, tWkvb + (long)l * RKVc * KVW, kv, Tc, KVW, RKVc, KVA, KVW, 0);

    // rope(q_pe) in place; build Keff = [kn | rope(k_pe)] per head
    {
      int tot = Bc * Sc * Hc * (DRc / 2);
      rope_q_kernel<<<(tot + 255) / 256, 256, 0, stream>>>(qb, rope_cos, rope_sin, tot);
      int tot2 = Bc * Sc * Hc * QHc;
      build_keff_kernel<<<(tot2 + 255) / 256, 256, 0, stream>>>(kv, kva, rope_cos,
                                                                rope_sin, keff, tot2);
    }

    // scores(b,h) = Q(b,h) [SxQH] @ Keff(b,h)^T
    launch_gemm_f32B(stream, qb, keff, sc, Sc, Sc, QHc,
                     QW, QW, Sc, /*trB=*/1, Bc * Hc, Hc,
                     (long)Sc * QW, QHc,
                     (long)Sc * QW, QHc,
                     (long)Hc * Sc * Sc, (long)Sc * Sc);

    softmax_kernel<<<Bc * Hc * Sc, Sc, 0, stream>>>(sc);

    // attn(b,h) = probs [SxS] @ V(b,h) [Sx64]
    launch_gemm_f32B(stream, sc, kv + DNc, ao, Sc, DVc, Sc,
                     Sc, KVW, Hc * DVc, /*trB=*/0, Bc * Hc, Hc,
                     (long)Hc * Sc * Sc, (long)Sc * Sc,
                     (long)Sc * KVW, KVHc,
                     (long)Sc * Hc * DVc, DVc);

    // hs += attn @ Wo
    launch_gemm_wt(stream, ao, tWo + (long)l * Dc * Dc, hs, Tc, Dc, Hc * DVc,
                   Hc * DVc, Dc, /*acc=*/1);

    // ---- MoE ----
    rms_kernel<<<Tc, 256, 0, stream>>>(hs, moe_nw + (long)l * Dc, h, Dc, Dc, Dc);

    matvec_small<<<Tc, 256, 0, stream>>>(h, Wr + (long)l * Dc * Ec, rlog, Ec, Dc, Dc, Ec, Ec);
    router_kernel<<<(Tc + 255) / 256, 256, 0, stream>>>(rlog, gate, Tc);

    for (int e = 0; e < Ec; ++e) {
      const _Float16* we1 = tWe1 + ((long)l * Ec + e) * Dc * Mc;
      const _Float16* we3 = tWe3 + ((long)l * Ec + e) * Dc * Mc;
      const _Float16* we2 = tWe2 + ((long)l * Ec + e) * Mc * Dc;
      launch_gemm_wt(stream, h, we1, g1, Tc, Mc, Dc, Dc, Mc, 0);
      launch_gemm_wt(stream, h, we3, g3, Tc, Mc, Dc, Dc, Mc, 0);
      int tot = Tc * Mc;
      silu_mul_kernel<<<(tot + 255) / 256, 256, 0, stream>>>(g1, g3, u, gate, e, Mc, tot);
      launch_gemm_wt(stream, u, we2, hs, Tc, Dc, Mc, Mc, Dc, /*acc=*/1);
    }

    // shared expert
    {
      const _Float16* ws1 = tWs1 + (long)l * Dc * MSc;
      const _Float16* ws3 = tWs3 + (long)l * Dc * MSc;
      const _Float16* ws2 = tWs2 + (long)l * MSc * Dc;
      launch_gemm_wt(stream, h, ws1, g1, Tc, MSc, Dc, Dc, MSc, 0);
      launch_gemm_wt(stream, h, ws3, g3, Tc, MSc, Dc, Dc, MSc, 0);
      int tot = Tc * MSc;
      silu_mul_kernel<<<(tot + 255) / 256, 256, 0, stream>>>(g1, g3, u, nullptr, 0, MSc, tot);
      launch_gemm_wt(stream, u, ws2, hs, Tc, Dc, MSc, MSc, Dc, /*acc=*/1);
    }
  }

  // final norm + head
  rms_kernel<<<Tc, 256, 0, stream>>>(hs, final_nw, h, Dc, Dc, Dc);
  matvec_small<<<Tc, 256, 0, stream>>>(h, W_out, out, 1, Dc, Dc, 1, 1);
}
